// GATMHAEfficient_816043786518
// MI455X (gfx1250) — matrix operand order repaired and checked by reading, not compiled, not run on hardware
//
#include <hip/hip_runtime.h>
#include <hip/hip_bf16.h>
#include <stdint.h>

typedef __attribute__((ext_vector_type(16))) _Float16 v16h;
typedef __attribute__((ext_vector_type(8)))  float    v8f;

#define H_DIM 8
#define B_DIM 32
#define N_DIM 1024
#define I_DIM 128
#define D_DIM 16
#define NEG_SLOPE 0.2f

// ---------------------------------------------------------------------------
// Kernel 1: per-head projection g = h @ W_h, plus a_l = g.Wal, a_r = g.War.
// Stores Gt transposed as (h, b, d, n) in f16 for WMMA B-operand loads.
// Block: 256 threads = 16 rows (n) x 16 cols (d). Grid: (N/16, B, H).
// ---------------------------------------------------------------------------
__global__ __launch_bounds__(256) void proj_kernel(
    const float* __restrict__ hbuf, const float* __restrict__ W,
    const float* __restrict__ Wal, const float* __restrict__ War,
    _Float16* __restrict__ Gt, float* __restrict__ aL, float* __restrict__ aR)
{
    __shared__ float sh[16][I_DIM];      // 8 KB: 16 h rows
    __shared__ float sw[I_DIM][D_DIM];   // 8 KB: W_h
    const int n0  = blockIdx.x * 16;
    const int b   = blockIdx.y;
    const int hh  = blockIdx.z;
    const int tid = threadIdx.x;

    const float4* hsrc = (const float4*)(hbuf + ((size_t)b * N_DIM + n0) * I_DIM);
    float4* hdst = (float4*)&sh[0][0];
    hdst[tid]       = hsrc[tid];
    hdst[tid + 256] = hsrc[tid + 256];
    const float4* wsrc = (const float4*)(W + (size_t)hh * I_DIM * D_DIM);
    float4* wdst = (float4*)&sw[0][0];
    wdst[tid]       = wsrc[tid];
    wdst[tid + 256] = wsrc[tid + 256];
    __syncthreads();

    const int d = tid & 15, r = tid >> 4;
    float acc = 0.f;
#pragma unroll 8
    for (int i = 0; i < I_DIM; ++i) acc += sh[r][i] * sw[i][d];

    const size_t hb = (size_t)hh * B_DIM + b;
    Gt[(hb * D_DIM + d) * N_DIM + n0 + r] = (_Float16)acc;

    float pl = acc * Wal[hh * D_DIM + d];
    float pr = acc * War[hh * D_DIM + d];
#pragma unroll
    for (int m = 8; m >= 1; m >>= 1) {
        pl += __shfl_xor(pl, m);
        pr += __shfl_xor(pr, m);
    }
    if (d == 0) {
        aL[hb * N_DIM + n0 + r] = pl;
        aR[hb * N_DIM + n0 + r] = pr;
    }
}

// ---------------------------------------------------------------------------
// Kernel 2: per-(h,b) reduce max over n of a_r (for the softmax max bound).
// ---------------------------------------------------------------------------
__global__ __launch_bounds__(256) void maxar_kernel(const float* __restrict__ aR,
                                                    float* __restrict__ maxAr)
{
    __shared__ float red[256];
    const int hb = blockIdx.x, tid = threadIdx.x;
    const float* p = aR + (size_t)hb * N_DIM;
    float m = fmaxf(fmaxf(p[tid], p[tid + 256]), fmaxf(p[tid + 512], p[tid + 768]));
    red[tid] = m;
    __syncthreads();
#pragma unroll
    for (int s = 128; s >= 1; s >>= 1) {
        if (tid < s) red[tid] = fmaxf(red[tid], red[tid + s]);
        __syncthreads();
    }
    if (tid == 0) maxAr[hb] = red[0];
}

// ---------------------------------------------------------------------------
// Kernel 3: fused masked-softmax + P@G via v_wmma_f32_16x16x32_f16.
// Grid: (8 tile-groups, H*B). Block 256 = 8 waves; each wave owns a 16-row
// tile and loops j in chunks of 32 (the WMMA K).
// Score P values are generated directly in the f16 A-matrix lane layout:
//   lane l: row r = l&15; half t -> K(t) = t + (t>=8)*8 + (l>=16)*8.
// B operand streams from pre-transposed Gt (two contiguous 16B runs per lane/chunk).
// Normalization uses one fast v_rcp_f32 per lane + bpermute of the
// reciprocal (instead of 8 IEEE division chains).
// ---------------------------------------------------------------------------
__global__ __launch_bounds__(256) void attn_kernel(
    const _Float16* __restrict__ Gt, const float* __restrict__ aL,
    const float* __restrict__ aR, const float* __restrict__ maxAr,
    const uint8_t* __restrict__ mask, float* __restrict__ out)
{
    __shared__ float s_ar[N_DIM];   // 4 KB: a_r for this (h,b)

    const int hb = blockIdx.y;
    const int hh = hb / B_DIM;
    const int b  = hb - hh * B_DIM;
    const int tid = threadIdx.x;

    // stage a_r into LDS (shared by all 8 waves)
    {
        const float4* src = (const float4*)(aR + (size_t)hb * N_DIM);
        ((float4*)s_ar)[tid] = src[tid];
    }
    __syncthreads();

    const int wave = tid >> 5;
    const int lane = tid & 31;
    const int r    = lane & 15;          // row within the 16-row tile
    const int hi   = lane >> 4;          // 0 or 1: lane group
    const int i0   = (blockIdx.x * 8 + wave) * 16;

    const float ai   = aL[(size_t)hb * N_DIM + i0 + r];
    const float mAr  = maxAr[hb];
    float       mrow = ai + mAr;
    mrow = fmaxf(mrow, NEG_SLOPE * mrow);        // leakyrelu (upper bound for row)

    const _Float16* gtp = Gt + ((size_t)hb * D_DIM + (lane & 15)) * N_DIM + hi * 8;
    const uint8_t* mrowp = mask + ((size_t)b * N_DIM + i0 + r) * N_DIM + hi * 8;
    const float4* ar4 = (const float4*)s_ar;

    v8f c = {};
    float suml = 0.f;

    for (int j0 = 0; j0 < N_DIM; j0 += 32) {
        // mask bytes: two contiguous 8B runs (K = base..base+7 and 16+base..+7)
        const uint2 mq0 = *(const uint2*)(mrowp + j0);
        const uint2 mq1 = *(const uint2*)(mrowp + j0 + 16);
        const uint64_t mb0 = ((uint64_t)mq0.y << 32) | mq0.x;
        const uint64_t mb1 = ((uint64_t)mq1.y << 32) | mq1.x;

        // a_r values for this lane's K set: two runs of 8 consecutive floats
        const int ia = (j0 + hi * 8) >> 2;
        const int ib = (j0 + 16 + hi * 8) >> 2;
        const float4 q0 = ar4[ia], q1 = ar4[ia + 1];
        const float4 q2 = ar4[ib], q3 = ar4[ib + 1];
        const float aj[16] = { q0.x, q0.y, q0.z, q0.w, q1.x, q1.y, q1.z, q1.w,
                               q2.x, q2.y, q2.z, q2.w, q3.x, q3.y, q3.z, q3.w };

        v16h a;
#pragma unroll
        for (int t = 0; t < 16; ++t) {
            float e = ai + aj[t];
            e = fmaxf(e, NEG_SLOPE * e);                       // leaky_relu
            const unsigned mbyte = (t < 8)
                ? (unsigned)((mb0 >> (8 * t)) & 0xff)
                : (unsigned)((mb1 >> (8 * (t - 8))) & 0xff);
            const float p = mbyte ? 0.f : __expf(e - mrow);    // in [0,1]
            suml += p;
            a[t] = (_Float16)p;
        }

        v16h bmat;
#pragma unroll
        for (int s = 0; s < 8; ++s) {
            bmat[s]     = gtp[j0 + s];
            bmat[s + 8] = gtp[j0 + 16 + s];
        }
        c = __builtin_amdgcn_wmma_f32_16x16x32_f16(
            /*neg_a=*/false, a, /*neg_b=*/false, bmat,
            /*c_mod=*/(short)0, c, /*reuse_a=*/false, /*reuse_b=*/false);
    }

    // row sums: lanes l and l^16 both accumulated row (l&15), disjoint K sets
    const float s2   = suml + __shfl_xor(suml, 16);      // lane l -> rowsum(l&15)
    const float rinv = __builtin_amdgcn_rcpf(s2);        // fast v_rcp_f32

    // C layout: vgpr p -> row M = p + 8*hi, column n = lane&15
#pragma unroll
    for (int p = 0; p < 8; ++p) {
        const int M = p + hi * 8;
        const float rs = __shfl(rinv, M);           // 1/rowsum(M) lives in lane M
        const float val = fmaxf(c[p] * rs, 0.f);    // softmax-normalize + relu
        out[(((size_t)b * N_DIM + i0 + M) * H_DIM + hh) * D_DIM + (lane & 15)] = val;
    }
}

// ---------------------------------------------------------------------------
extern "C" void kernel_launch(void* const* d_in, const int* in_sizes, int n_in,
                              void* d_out, int out_size, void* d_ws, size_t ws_size,
                              hipStream_t stream) {
    const float*   h_in = (const float*)d_in[0];   // (B,N,I)
    const float*   W    = (const float*)d_in[1];   // (H,I,D)
    const float*   Wal  = (const float*)d_in[2];   // (H,D,1)
    const float*   War  = (const float*)d_in[3];   // (H,D,1)
    const uint8_t* mask = (const uint8_t*)d_in[4]; // (B,N,N) bool
    float* out = (float*)d_out;                    // (B,N,H*D)

    char* ws = (char*)d_ws;
    _Float16* Gt    = (_Float16*)(ws);                    //  8,388,608 B
    float*    aL    = (float*)   (ws +  8388608);         //  1,048,576 B
    float*    aR    = (float*)   (ws +  9437184);         //  1,048,576 B
    float*    maxAr = (float*)   (ws + 10485760);         //      1,024 B

    proj_kernel<<<dim3(N_DIM / 16, B_DIM, H_DIM), 256, 0, stream>>>(
        h_in, W, Wal, War, Gt, aL, aR);
    maxar_kernel<<<H_DIM * B_DIM, 256, 0, stream>>>(aR, maxAr);
    attn_kernel<<<dim3(8, H_DIM * B_DIM), 256, 0, stream>>>(
        Gt, aL, aR, maxAr, mask, out);
}
